// NutriGraphNetEncoder_11098195493027
// MI455X (gfx1250) — matrix operand order repaired and checked by reading, not compiled, not run on hardware
//
#include <hip/hip_runtime.h>
#include <hip/hip_bf16.h>

// GAT hetero-GNN forward for MI455X (gfx1250, wave32).
// Dense projections use v_wmma_f32_16x16x32_f16; edge softmax/scatter use
// f32 global atomics with a wave-per-edge float4 message scatter and
// global_prefetch_b8 software prefetch of the next message row.

typedef __attribute__((ext_vector_type(16))) _Float16 v16h;
typedef __attribute__((ext_vector_type(8)))  _Float16 v8h;
typedef __attribute__((ext_vector_type(8)))  float    v8f;

#define CDIV(a,b) (((a)+(b)-1)/(b))

// ---------------------------------------------------------------------------
// C[M,N] = A[M,K] (f16, row-major) x Wt[N,K] (f16, pre-transposed weight)
// One wave per 16x16 output tile; blockDim.x = 32*(N/16); gridDim.x = M/16.
// Lane layouts per CDNA5 ISA 7.12.2:
//   A (16x32 f16): lane<16 -> row=lane, K = k0+{0..7, 16..23}
//                  lane>=16 -> row=lane-16, K = k0+{8..15, 24..31}
//   B (32x16 f16): lane<16 -> col=lane, K = k0+{0..7, 16..23}; lane>=16 -> K = k0+{8..15, 24..31}
//   C/D (16x16 f32): elem r -> row r + 8*(lane>=16), col = lane&15
// ---------------------------------------------------------------------------
__global__ void gemm_wmma_kernel(const _Float16* __restrict__ A,
                                 const _Float16* __restrict__ Wt,
                                 float* __restrict__ C,
                                 int M, int K, int N) {
  const int wave = threadIdx.x >> 5;
  const int lane = threadIdx.x & 31;
  const int hi   = lane >> 4;
  const int lr   = lane & 15;
  const int m0   = blockIdx.x << 4;
  const int n0   = wave << 4;

  int arow = m0 + lr; if (arow >= M) arow = M - 1;
  const _Float16* Ar = A  + (size_t)arow * K;
  const _Float16* Br = Wt + (size_t)(n0 + lr) * K;

  v8f acc = {};
  for (int k0 = 0; k0 < K; k0 += 32) {
    const int ka = k0 + (hi ? 8 : 0);
    v8h alo = *(const v8h*)(Ar + ka);
    v8h ahi = *(const v8h*)(Ar + ka + 16);
    v16h a;
#pragma unroll
    for (int i = 0; i < 8; ++i) { a[i] = alo[i]; a[i + 8] = ahi[i]; }
    v8h blo = *(const v8h*)(Br + ka);
    v8h bhi = *(const v8h*)(Br + ka + 16);
    v16h b;
#pragma unroll
    for (int i = 0; i < 8; ++i) { b[i] = blo[i]; b[i + 8] = bhi[i]; }
    acc = __builtin_amdgcn_wmma_f32_16x16x32_f16(false, a, false, b,
                                                 (short)0, acc, false, false);
  }
#pragma unroll
  for (int r = 0; r < 8; ++r) {
    const int row = m0 + r + (hi << 3);
    if (row < M) C[(size_t)row * N + n0 + lr] = acc[r];
  }
}

// ---------------------------------------------------------------------------
// Elementwise / conversion helpers
// ---------------------------------------------------------------------------
__global__ void cvt_f16_kernel(const float* __restrict__ in,
                               _Float16* __restrict__ out, int n) {
  int i = blockIdx.x * blockDim.x + threadIdx.x;
  if (i < n) out[i] = (_Float16)in[i];
}

// W[K,N] f32 -> Wt[N,K] f16
__global__ void cvt_wt_kernel(const float* __restrict__ W,
                              _Float16* __restrict__ Wt, int K, int N) {
  int i = blockIdx.x * blockDim.x + threadIdx.x;
  if (i < K * N) {
    int k = i / N, n = i - k * N;
    Wt[(size_t)n * K + k] = (_Float16)W[i];
  }
}

__global__ void zero_f32_kernel(float* __restrict__ p, int n) {
  int i = blockIdx.x * blockDim.x + threadIdx.x;
  if (i < n) p[i] = 0.0f;
}

// c = dot(W_edge[0,:], a_edge)  (edge_dim == 1)
__global__ void edge_coef_kernel(const float* __restrict__ We,
                                 const float* __restrict__ ae,
                                 float* __restrict__ c, int d) {
  if (blockIdx.x == 0 && threadIdx.x == 0) {
    float s = 0.0f;
    for (int j = 0; j < d; ++j) s += We[j] * ae[j];
    *c = s;
  }
}

// s[node] = dot(X[node, 0:128], a[0:128]); one wave32 per node.
__global__ void rowdot_kernel(const float* __restrict__ X,
                              const float* __restrict__ a,
                              float* __restrict__ out, int n) {
  int node = (blockIdx.x * blockDim.x + threadIdx.x) >> 5;
  int lane = threadIdx.x & 31;
  if (node >= n) return;
  float4 v = ((const float4*)(X + (size_t)node * 128))[lane];
  float4 w = ((const float4*)a)[lane];
  float s = v.x * w.x + v.y * w.y + v.z * w.z + v.w * w.w;
  for (int o = 16; o; o >>= 1) s += __shfl_xor(s, o, 32);
  if (lane == 0) out[node] = s;
}

// ---------------------------------------------------------------------------
// Edge-softmax pipeline (monotonic-uint float max encoding)
// ---------------------------------------------------------------------------
__device__ __forceinline__ unsigned fenc(float f) {
  unsigned u = __float_as_uint(f);
  return (u >> 31) ? ~u : (u | 0x80000000u);
}
__device__ __forceinline__ float fdec(unsigned k) {
  return __uint_as_float((k & 0x80000000u) ? (k ^ 0x80000000u) : ~k);
}

__global__ void edge_logits_kernel(const int* __restrict__ src,
                                   const int* __restrict__ dst,
                                   const float* __restrict__ ea,   // nullable [E]
                                   const float* __restrict__ ce,   // nullable scalar
                                   const float* __restrict__ ssrc,
                                   const float* __restrict__ sdst,
                                   float* __restrict__ evals,
                                   unsigned* __restrict__ menc, int E) {
  int e = blockIdx.x * blockDim.x + threadIdx.x;
  if (e >= E) return;
  float v = ssrc[src[e]] + sdst[dst[e]];
  if (ea != nullptr) v += ea[e] * (*ce);
  v = (v > 0.0f) ? v : 0.2f * v;  // leaky_relu, slope 0.2
  evals[e] = v;
  atomicMax(&menc[dst[e]], fenc(v));
}

__global__ void edge_expsum_kernel(const int* __restrict__ dst,
                                   float* __restrict__ evals,
                                   const unsigned* __restrict__ menc,
                                   float* __restrict__ ssum, int E) {
  int e = blockIdx.x * blockDim.x + threadIdx.x;
  if (e >= E) return;
  int d = dst[e];
  float w = __expf(evals[e] - fdec(menc[d]));
  evals[e] = w;
  atomicAdd(&ssum[d], w);
}

// Grid-stride loop, one wave32 per edge per iteration: lane moves one float4
// (128-f32 message). The next iteration's source row is software-prefetched
// (global_prefetch_b8) so the gather latency hides behind the atomic drain.
__global__ void edge_scatter_kernel(const int* __restrict__ src,
                                    const int* __restrict__ dst,
                                    const float* __restrict__ evals,
                                    const float* __restrict__ ssum,
                                    const float* __restrict__ hs,
                                    float* __restrict__ out, int E) {
  const int lane   = threadIdx.x & 31;
  const int stride = (gridDim.x * blockDim.x) >> 5;  // total waves
  int e = (blockIdx.x * blockDim.x + threadIdx.x) >> 5;
  for (; e < E; e += stride) {
    const int en = e + stride;
    if (en < E && lane == 0) {
      __builtin_prefetch(hs + (size_t)src[en] * 128, 0, 0);
    }
    const int s = src[e], d = dst[e];
    const float alpha = evals[e] / (ssum[d] + 1e-16f);
    const float4 v = ((const float4*)(hs + (size_t)s * 128))[lane];
    float* o = out + (size_t)d * 128 + lane * 4;
    atomicAdd(o + 0, alpha * v.x);
    atomicAdd(o + 1, alpha * v.y);
    atomicAdd(o + 2, alpha * v.z);
    atomicAdd(o + 3, alpha * v.w);
  }
}

// out = ELU(BN(acc + b)); optional f16 mirror for next-layer WMMA input.
__global__ void bias_bn_elu_kernel(const float* __restrict__ acc,
                                   const float* __restrict__ b,
                                   const float* __restrict__ gamma,
                                   const float* __restrict__ beta,
                                   const float* __restrict__ mean,
                                   const float* __restrict__ var,
                                   float* __restrict__ outf,
                                   _Float16* __restrict__ outh, int n) {
  int i = blockIdx.x * blockDim.x + threadIdx.x;
  if (i >= n * 128) return;
  int d = i & 127;
  float y = acc[i] + b[d];
  y = (y - mean[d]) * rsqrtf(var[d] + 1e-5f) * gamma[d] + beta[d];
  y = (y > 0.0f) ? y : (__expf(y) - 1.0f);
  outf[i] = y;
  if (outh != nullptr) outh[i] = (_Float16)y;
}

// pref = sigmoid(relu(h1 + bh1) @ Wh2 + bh2), h1: [n,64]
__global__ void pref_head_kernel(const float* __restrict__ h1,
                                 const float* __restrict__ bh1,
                                 const float* __restrict__ Wh2,
                                 const float* __restrict__ bh2,
                                 float* __restrict__ pref, int n) {
  int i = blockIdx.x * blockDim.x + threadIdx.x;
  if (i >= n) return;
  const float* r = h1 + (size_t)i * 64;
  float s = 0.0f;
  for (int j = 0; j < 64; ++j) {
    float v = r[j] + bh1[j];
    v = (v > 0.0f) ? v : 0.0f;
    s += v * Wh2[j];
  }
  s += bh2[0];
  pref[i] = 1.0f / (1.0f + __expf(-s));
}

__global__ void health_adj_kernel(const int* __restrict__ hsrc,
                                  const int* __restrict__ hdst,
                                  const float* __restrict__ hsc,
                                  const float* __restrict__ pref,
                                  float* __restrict__ upd, int E) {
  int e = blockIdx.x * blockDim.x + threadIdx.x;
  if (e >= E) return;
  atomicAdd(&upd[hdst[e]], pref[hsrc[e]] * hsc[e]);
}

__global__ void apply_upd_kernel(float* __restrict__ f2,
                                 const float* __restrict__ upd, int n) {
  int i = blockIdx.x * blockDim.x + threadIdx.x;
  if (i >= n * 128) return;
  f2[i] += 0.1f * upd[i >> 7];
}

// ---------------------------------------------------------------------------
// Host orchestration
// ---------------------------------------------------------------------------
extern "C" void kernel_launch(void* const* d_in, const int* in_sizes, int n_in,
                              void* d_out, int out_size, void* d_ws, size_t ws_size,
                              hipStream_t stream) {
  // ---- raw tensor inputs (setup_inputs() dict order) ----
  const float* x_user   = (const float*)d_in[0];   // [NU,128]
  const float* x_food   = (const float*)d_in[1];   // [NF,64]
  const int*   e_uf_src = (const int*)  d_in[2];
  const int*   e_uf_dst = (const int*)  d_in[3];
  const float* ea_uf    = (const float*)d_in[4];   // [E_UF,1]
  const int*   e_fu_src = (const int*)  d_in[5];
  const int*   e_fu_dst = (const int*)  d_in[6];
  const int*   h_src    = (const int*)  d_in[7];
  const int*   h_dst    = (const int*)  d_in[8];
  const float* h_score  = (const float*)d_in[9];

  const int NU   = in_sizes[0] / 128;
  const int NF   = in_sizes[1] / 64;
  const int E_UF = in_sizes[2];
  const int E_FU = in_sizes[5];
  const int E_H  = in_sizes[7];

  // ---- params, assumed JAX-pytree (recursively key-sorted) flatten order ----
  // Wh1, Wh2, bh1, bh2,
  // bn1_food{beta,gamma,mean,var}, bn1_user{..}, bn2_food{..}, bn2_user{..},
  // l1_fu{W_dst,W_src,a_dst,a_src,b},
  // l1_uf{W_dst,W_edge,W_src,a_dst,a_edge,a_src,b},
  // l2_fu{W_dst,W_src,a_dst,a_src,b},
  // l2_uf{W_dst,W_edge,W_src,a_dst,a_edge,a_src,b}
  int p = 10;
  const float* Wh1 = (const float*)d_in[p++];      // [128,64]
  const float* Wh2 = (const float*)d_in[p++];      // [64,1]
  const float* bh1 = (const float*)d_in[p++];
  const float* bh2 = (const float*)d_in[p++];
  const float* bn1f[4]; for (int i = 0; i < 4; ++i) bn1f[i] = (const float*)d_in[p++];
  const float* bn1u[4]; for (int i = 0; i < 4; ++i) bn1u[i] = (const float*)d_in[p++];
  const float* bn2f[4]; for (int i = 0; i < 4; ++i) bn2f[i] = (const float*)d_in[p++];
  const float* bn2u[4]; for (int i = 0; i < 4; ++i) bn2u[i] = (const float*)d_in[p++];
  // bn order within each: beta, gamma, mean, var
  const float* l1fu_Wdst = (const float*)d_in[p++]; // [128,128]
  const float* l1fu_Wsrc = (const float*)d_in[p++]; // [64,128]
  const float* l1fu_adst = (const float*)d_in[p++];
  const float* l1fu_asrc = (const float*)d_in[p++];
  const float* l1fu_b    = (const float*)d_in[p++];
  const float* l1uf_Wdst = (const float*)d_in[p++]; // [64,128]
  const float* l1uf_Wedg = (const float*)d_in[p++]; // [1,128]
  const float* l1uf_Wsrc = (const float*)d_in[p++]; // [128,128]
  const float* l1uf_adst = (const float*)d_in[p++];
  const float* l1uf_aedg = (const float*)d_in[p++];
  const float* l1uf_asrc = (const float*)d_in[p++];
  const float* l1uf_b    = (const float*)d_in[p++];
  const float* l2fu_Wdst = (const float*)d_in[p++]; // [128,128]
  const float* l2fu_Wsrc = (const float*)d_in[p++]; // [128,128]
  const float* l2fu_adst = (const float*)d_in[p++];
  const float* l2fu_asrc = (const float*)d_in[p++];
  const float* l2fu_b    = (const float*)d_in[p++];
  const float* l2uf_Wdst = (const float*)d_in[p++];
  const float* l2uf_Wedg = (const float*)d_in[p++];
  const float* l2uf_Wsrc = (const float*)d_in[p++];
  const float* l2uf_adst = (const float*)d_in[p++];
  const float* l2uf_aedg = (const float*)d_in[p++];
  const float* l2uf_asrc = (const float*)d_in[p++];
  const float* l2uf_b    = (const float*)d_in[p++];

  // ---- workspace bump allocator ----
  char* wsp = (char*)d_ws;
  auto alloc = [&](size_t bytes) -> void* {
    void* q = wsp; wsp += (bytes + 255) & ~(size_t)255; return q;
  };

  _Float16* xu_h = (_Float16*)alloc((size_t)NU * 128 * 2);
  _Float16* xf_h = (_Float16*)alloc((size_t)NF * 64 * 2);
  _Float16* u1_h = (_Float16*)alloc((size_t)NU * 128 * 2);
  _Float16* f1_h = (_Float16*)alloc((size_t)NF * 128 * 2);
  _Float16* u2_h = (_Float16*)alloc((size_t)NU * 128 * 2);

  _Float16* t_l1uf_Wsrc = (_Float16*)alloc(128 * 128 * 2); // [N=128,K=128]
  _Float16* t_l1uf_Wdst = (_Float16*)alloc(128 * 64 * 2);  // [N=128,K=64]
  _Float16* t_l1fu_Wsrc = (_Float16*)alloc(128 * 64 * 2);
  _Float16* t_l1fu_Wdst = (_Float16*)alloc(128 * 128 * 2);
  _Float16* t_l2uf_Wsrc = (_Float16*)alloc(128 * 128 * 2);
  _Float16* t_l2uf_Wdst = (_Float16*)alloc(128 * 128 * 2);
  _Float16* t_l2fu_Wsrc = (_Float16*)alloc(128 * 128 * 2);
  _Float16* t_l2fu_Wdst = (_Float16*)alloc(128 * 128 * 2);
  _Float16* t_Wh1       = (_Float16*)alloc(64 * 128 * 2);  // [N=64,K=128]
  float* ce1 = (float*)alloc(4);
  float* ce2 = (float*)alloc(4);

  float* u1 = (float*)alloc((size_t)NU * 128 * 4);
  float* f1 = (float*)alloc((size_t)NF * 128 * 4);

  float*    hs    = (float*)alloc((size_t)NU * 128 * 4);  // also reused as h1 [NU,64]
  float*    hd    = (float*)alloc((size_t)NU * 128 * 4);  // also reused as accum
  float*    ssrc  = (float*)alloc((size_t)NU * 4);
  float*    sdst  = (float*)alloc((size_t)NU * 4);        // also reused as upd
  unsigned* menc  = (unsigned*)alloc((size_t)NU * 4);
  float*    ssum  = (float*)alloc((size_t)NU * 4);
  float*    evals = (float*)alloc((size_t)(E_UF > E_FU ? E_UF : E_FU) * 4);

  float* out_u2   = (float*)d_out;
  float* out_f2   = out_u2 + (size_t)NU * 128;
  float* out_pref = out_f2 + (size_t)NF * 128;

  const int TB = 256;
  const int SCATTER_BLKS = 8192;  // grid-stride scatter: 65536 waves resident

  // ---- prep: f16 conversions, weight transposes, edge coefficients ----
  cvt_f16_kernel<<<CDIV(NU * 128, TB), TB, 0, stream>>>(x_user, xu_h, NU * 128);
  cvt_f16_kernel<<<CDIV(NF * 64, TB),  TB, 0, stream>>>(x_food, xf_h, NF * 64);
  cvt_wt_kernel<<<CDIV(128 * 128, TB), TB, 0, stream>>>(l1uf_Wsrc, t_l1uf_Wsrc, 128, 128);
  cvt_wt_kernel<<<CDIV(64 * 128, TB),  TB, 0, stream>>>(l1uf_Wdst, t_l1uf_Wdst, 64, 128);
  cvt_wt_kernel<<<CDIV(64 * 128, TB),  TB, 0, stream>>>(l1fu_Wsrc, t_l1fu_Wsrc, 64, 128);
  cvt_wt_kernel<<<CDIV(128 * 128, TB), TB, 0, stream>>>(l1fu_Wdst, t_l1fu_Wdst, 128, 128);
  cvt_wt_kernel<<<CDIV(128 * 128, TB), TB, 0, stream>>>(l2uf_Wsrc, t_l2uf_Wsrc, 128, 128);
  cvt_wt_kernel<<<CDIV(128 * 128, TB), TB, 0, stream>>>(l2uf_Wdst, t_l2uf_Wdst, 128, 128);
  cvt_wt_kernel<<<CDIV(128 * 128, TB), TB, 0, stream>>>(l2fu_Wsrc, t_l2fu_Wsrc, 128, 128);
  cvt_wt_kernel<<<CDIV(128 * 128, TB), TB, 0, stream>>>(l2fu_Wdst, t_l2fu_Wdst, 128, 128);
  cvt_wt_kernel<<<CDIV(128 * 64, TB),  TB, 0, stream>>>(Wh1, t_Wh1, 128, 64);
  edge_coef_kernel<<<1, 1, 0, stream>>>(l1uf_Wedg, l1uf_aedg, ce1, 128);
  edge_coef_kernel<<<1, 1, 0, stream>>>(l2uf_Wedg, l2uf_aedg, ce2, 128);

  // ---- one GAT pass: out[nd,128] = ELU(BN(segment_softmax_aggregate + b)) ----
  auto run_gat = [&](const _Float16* xs_h, int ns, int ks, const _Float16* Wsrc_t,
                     const _Float16* xd_h, int nd, int kd, const _Float16* Wdst_t,
                     const float* a_src, const float* a_dst, const float* bvec,
                     const int* esrc, const int* edst, int E,
                     const float* eattr, const float* ce,
                     const float* const* bn,  // beta,gamma,mean,var
                     float* out_f32, _Float16* out_f16) {
    gemm_wmma_kernel<<<ns / 16, 256, 0, stream>>>(xs_h, Wsrc_t, hs, ns, ks, 128);
    gemm_wmma_kernel<<<nd / 16, 256, 0, stream>>>(xd_h, Wdst_t, hd, nd, kd, 128);
    rowdot_kernel<<<CDIV(ns * 32, TB), TB, 0, stream>>>(hs, a_src, ssrc, ns);
    rowdot_kernel<<<CDIV(nd * 32, TB), TB, 0, stream>>>(hd, a_dst, sdst, nd);
    // hd no longer needed -> reuse as the [nd,128] accumulator
    float* accum = hd;
    zero_f32_kernel<<<CDIV(nd * 128, TB), TB, 0, stream>>>(accum, nd * 128);
    zero_f32_kernel<<<CDIV(nd, TB), TB, 0, stream>>>((float*)menc, nd); // 0u == 0.0f bits
    zero_f32_kernel<<<CDIV(nd, TB), TB, 0, stream>>>(ssum, nd);
    edge_logits_kernel<<<CDIV(E, TB), TB, 0, stream>>>(esrc, edst, eattr, ce,
                                                       ssrc, sdst, evals, menc, E);
    edge_expsum_kernel<<<CDIV(E, TB), TB, 0, stream>>>(edst, evals, menc, ssum, E);
    edge_scatter_kernel<<<SCATTER_BLKS, TB, 0, stream>>>(esrc, edst, evals, ssum,
                                                         hs, accum, E);
    bias_bn_elu_kernel<<<CDIV(nd * 128, TB), TB, 0, stream>>>(
        accum, bvec, bn[1], bn[0], bn[2], bn[3], out_f32, out_f16, nd);
  };

  // ---- layer 1 ----
  run_gat(xu_h, NU, 128, t_l1uf_Wsrc, xf_h, NF, 64, t_l1uf_Wdst,
          l1uf_asrc, l1uf_adst, l1uf_b, e_uf_src, e_uf_dst, E_UF, ea_uf, ce1,
          bn1f, f1, f1_h);
  run_gat(xf_h, NF, 64, t_l1fu_Wsrc, xu_h, NU, 128, t_l1fu_Wdst,
          l1fu_asrc, l1fu_adst, l1fu_b, e_fu_src, e_fu_dst, E_FU, nullptr, nullptr,
          bn1u, u1, u1_h);

  // ---- layer 2 (writes directly into d_out) ----
  run_gat(u1_h, NU, 128, t_l2uf_Wsrc, f1_h, NF, 128, t_l2uf_Wdst,
          l2uf_asrc, l2uf_adst, l2uf_b, e_uf_src, e_uf_dst, E_UF, ea_uf, ce2,
          bn2f, out_f2, nullptr);
  run_gat(f1_h, NF, 128, t_l2fu_Wsrc, u1_h, NU, 128, t_l2fu_Wdst,
          l2fu_asrc, l2fu_adst, l2fu_b, e_fu_src, e_fu_dst, E_FU, nullptr, nullptr,
          bn2u, out_u2, u2_h);

  // ---- preference head: h1 = u2 @ Wh1 (WMMA, N=64), then relu/Wh2/sigmoid ----
  float* h1buf = hs;  // [NU,64] fits in hs scratch
  gemm_wmma_kernel<<<NU / 16, 128, 0, stream>>>(u2_h, t_Wh1, h1buf, NU, 128, 64);
  pref_head_kernel<<<CDIV(NU, TB), TB, 0, stream>>>(h1buf, bh1, Wh2, bh2, out_pref, NU);

  // ---- health adjustment: f2 += 0.1 * segment_sum(pref[src]*score, dst) ----
  float* upd = sdst;  // [NF]
  zero_f32_kernel<<<CDIV(NF, TB), TB, 0, stream>>>(upd, NF);
  health_adj_kernel<<<CDIV(E_H, TB), TB, 0, stream>>>(h_src, h_dst, h_score,
                                                      out_pref, upd, E_H);
  apply_upd_kernel<<<CDIV(NF * 128, TB), TB, 0, stream>>>(out_f2, upd, NF);

  (void)n_in; (void)out_size; (void)ws_size;
}
